// Conan_10522669875840
// MI455X (gfx1250) — matrix operand order repaired and checked by reading, not compiled, not run on hardware
//
#include <hip/hip_runtime.h>
#include <hip/hip_bf16.h>

typedef __attribute__((ext_vector_type(16))) __bf16 v16bf;
typedef __attribute__((ext_vector_type(8)))  float  v8f;
typedef __attribute__((ext_vector_type(4)))  float  v4f;

#define BATCH 8
#define NTILE 20000
#define DIM   1024
#define H1    32
#define H2    8
#define HF    32
#define KSEL  10
#define FEAT  ((H1 + 1) * 2 * KSEL + H1)   // 692
#define ROWS_PER_BLOCK 128                  // 8 waves * 16 rows

// ---------------------------------------------------------------------------
// Stage 1: per-tile MLP. Layer1 (1024->32) via bf16 WMMA w/ f32 accum,
// software-pipelined one K-chunk ahead; layers 2..4 (32->8->32->1) per-row
// VALU. Writes h3 [rows,32] + scores [rows] to workspace.
// ---------------------------------------------------------------------------
__global__ __launch_bounds__(256) void conan_stage1(
    const float* __restrict__ x,
    const float* __restrict__ W1, const float* __restrict__ b1,
    const float* __restrict__ W2, const float* __restrict__ b2,
    const float* __restrict__ W3, const float* __restrict__ b3,
    const float* __restrict__ Wsc, const float* __restrict__ bsc,
    float* __restrict__ out_ws, float* __restrict__ scores_ws)
{
    __shared__ __bf16 w1_lds[H1 * DIM];                 // 64 KB, row-major [n][k]
    __shared__ float  h1_lds[ROWS_PER_BLOCK * H1];      // 16 KB

    const int tid = threadIdx.x;

    // Convert W1 fp32 -> bf16 into LDS once per workgroup (L2-resident source).
    for (int i = tid; i < H1 * DIM; i += 256)
        w1_lds[i] = (__bf16)W1[i];
    __syncthreads();

    const int wave = tid >> 5;
    const int lane = tid & 31;
    const int hl   = lane >> 4;     // lane group (0: lanes 0-15, 1: lanes 16-31)
    const int lr   = lane & 15;

    const long rowBase = (long)blockIdx.x * ROWS_PER_BLOCK + wave * 16;
    const float* xrow  = x + (rowBase + lr) * (long)DIM;
    const __bf16* w0p  = &w1_lds[lr * DIM + 16 * hl];        // B frag 0 base
    const __bf16* w1p  = &w1_lds[(16 + lr) * DIM + 16 * hl]; // B frag 1 base

    v8f c0 = {};   // output channels 0..15
    v8f c1 = {};   // output channels 16..31

    // ---- software pipeline prologue: chunk k0 = 0 in flight ----
    v4f f0 = *(const v4f*)(xrow + 16 * hl);
    v4f f1 = *(const v4f*)(xrow + 16 * hl + 4);
    v4f f2 = *(const v4f*)(xrow + 16 * hl + 8);
    v4f f3 = *(const v4f*)(xrow + 16 * hl + 12);
    v16bf B0 = *(const v16bf*)(w0p);
    v16bf B1 = *(const v16bf*)(w1p);

    for (int k0 = 0; k0 < DIM; k0 += 32) {
        const int kn = (k0 + 32) & (DIM - 1);   // wrapped lookahead: always in-bounds

        // Issue next chunk's loads first (consumed one iteration later).
        v4f g0 = *(const v4f*)(xrow + kn + 16 * hl);
        v4f g1 = *(const v4f*)(xrow + kn + 16 * hl + 4);
        v4f g2 = *(const v4f*)(xrow + kn + 16 * hl + 8);
        v4f g3 = *(const v4f*)(xrow + kn + 16 * hl + 12);
        v16bf Bn0 = *(const v16bf*)(w0p + kn);
        v16bf Bn1 = *(const v16bf*)(w1p + kn);
        __builtin_prefetch(xrow + ((k0 + 96) & (DIM - 1)), 0, 0);  // GL2 stream-ahead

        // Convert current chunk (loads landed last iteration) and multiply.
        // A layout (16x32 bf16): elems 0..15 = K k0+16*hl.. (contiguous, matches B)
        v16bf A;
#pragma unroll
        for (int i = 0; i < 4; ++i) {
            A[i]      = (__bf16)f0[i];
            A[4 + i]  = (__bf16)f1[i];
            A[8 + i]  = (__bf16)f2[i];
            A[12 + i] = (__bf16)f3[i];
        }
        c0 = __builtin_amdgcn_wmma_f32_16x16x32_bf16(false, A, false, B0,
                                                     (short)0, c0, false, false);
        c1 = __builtin_amdgcn_wmma_f32_16x16x32_bf16(false, A, false, B1,
                                                     (short)0, c1, false, false);

        // rotate pipeline registers
        f0 = g0; f1 = g1; f2 = g2; f3 = g3;
        B0 = Bn0; B1 = Bn1;
    }

    // Bias + ReLU, scatter C-layout (lane = channel, vgpr = row) into row-major LDS.
    const float bb0 = b1[lr];
    const float bb1 = b1[16 + lr];
#pragma unroll
    for (int i = 0; i < 8; ++i) {
        const int m = i + 8 * hl;               // row within the wave's 16-row tile
        float v0 = c0[i] + bb0;  v0 = v0 > 0.f ? v0 : 0.f;
        float v1 = c1[i] + bb1;  v1 = v1 > 0.f ? v1 : 0.f;
        h1_lds[(wave * 16 + m) * H1 + lr]      = v0;
        h1_lds[(wave * 16 + m) * H1 + 16 + lr] = v1;
    }
    __syncthreads();

    // Per-row tail MLP: 32->8->32->score (tiny; VALU). Threads 0..127 own one row each.
    if (tid < ROWS_PER_BLOCK) {
        const float* h1 = &h1_lds[tid * H1];
        float h2[H2];
#pragma unroll
        for (int g = 0; g < H2; ++g) {
            float acc = b2[g];
#pragma unroll
            for (int j = 0; j < H1; ++j) acc += W2[g * H1 + j] * h1[j];
            h2[g] = acc > 0.f ? acc : 0.f;
        }
        const long grow = (long)blockIdx.x * ROWS_PER_BLOCK + tid;
        float* orow = out_ws + grow * H1;
        float sacc = bsc[0];
#pragma unroll
        for (int h = 0; h < H1; ++h) {
            float acc = b3[h];
#pragma unroll
            for (int g = 0; g < H2; ++g) acc += W3[h * H2 + g] * h2[g];
            acc = acc > 0.f ? acc : 0.f;
            orow[h] = acc;
            sacc += Wsc[h] * acc;
        }
        scores_ws[grow] = sacc > 0.f ? sacc : 0.f;
    }
}

// ---------------------------------------------------------------------------
// Stage 2: per batch, bottom-10 (ascending) and top-10 (ascending, like
// argsort[-K:]) indices via 20 masked reduction passes over 20000 scores.
// ---------------------------------------------------------------------------
__global__ __launch_bounds__(256) void conan_select(
    const float* __restrict__ scores, int* __restrict__ sel)
{
    __shared__ float sval[256];
    __shared__ int   sidx[256];
    __shared__ int   ssel[2 * KSEL];

    const int b = blockIdx.x;
    const int t = threadIdx.x;
    const float* s = scores + (long)b * NTILE;

    // bottom-K: repeated min (tie -> smaller index, matching stable argsort)
    for (int p = 0; p < KSEL; ++p) {
        float best = 3.402823466e+38f; int bidx = NTILE;
        for (int i = t; i < NTILE; i += 256) {
            bool taken = false;
            for (int q = 0; q < p; ++q) taken |= (ssel[q] == i);
            const float v = s[i];
            if (!taken && (v < best || (v == best && i < bidx))) { best = v; bidx = i; }
        }
        sval[t] = best; sidx[t] = bidx; __syncthreads();
        for (int off = 128; off > 0; off >>= 1) {
            if (t < off) {
                const float v2 = sval[t + off]; const int i2 = sidx[t + off];
                if (v2 < sval[t] || (v2 == sval[t] && i2 < sidx[t])) { sval[t] = v2; sidx[t] = i2; }
            }
            __syncthreads();
        }
        if (t == 0) ssel[p] = sidx[0];
        __syncthreads();
    }
    // top-K: repeated max (tie -> larger index)
    for (int p = 0; p < KSEL; ++p) {
        float best = -3.402823466e+38f; int bidx = -1;
        for (int i = t; i < NTILE; i += 256) {
            bool taken = false;
            for (int q = 0; q < p; ++q) taken |= (ssel[KSEL + q] == i);
            const float v = s[i];
            if (!taken && (v > best || (v == best && i > bidx))) { best = v; bidx = i; }
        }
        sval[t] = best; sidx[t] = bidx; __syncthreads();
        for (int off = 128; off > 0; off >>= 1) {
            if (t < off) {
                const float v2 = sval[t + off]; const int i2 = sidx[t + off];
                if (v2 > sval[t] || (v2 == sval[t] && i2 > sidx[t])) { sval[t] = v2; sidx[t] = i2; }
            }
            __syncthreads();
        }
        if (t == 0) ssel[KSEL + p] = sidx[0];
        __syncthreads();
    }
    // sel layout: [0..9] = mins ascending; [10..19] = argsort[-K:] (10th-largest..largest)
    if (t < KSEL)           sel[b * 2 * KSEL + t] = ssel[t];
    else if (t < 2 * KSEL)  sel[b * 2 * KSEL + KSEL + (KSEL - 1 - (t - KSEL))] = ssel[t];
}

// ---------------------------------------------------------------------------
// Stage 3: gather feature vector [692] and run the tiny classifier head.
// ---------------------------------------------------------------------------
__global__ __launch_bounds__(128) void conan_classify(
    const float* __restrict__ out_ws, const float* __restrict__ scores_ws,
    const int* __restrict__ sel,
    const float* __restrict__ Wc1, const float* __restrict__ bc1,
    const float* __restrict__ Wc2, const float* __restrict__ bc2,
    const float* __restrict__ Wc3, const float* __restrict__ bc3,
    float* __restrict__ out)
{
    __shared__ float feat[FEAT];
    __shared__ int   ssel[2 * KSEL];
    __shared__ float z1[HF], z2[HF];

    const int b = blockIdx.x;
    const int t = threadIdx.x;

    if (t < 2 * KSEL) ssel[t] = sel[b * 2 * KSEL + t];
    __syncthreads();

    if (t < 2 * KSEL) feat[t] = scores_ws[(long)b * NTILE + ssel[t]];
    for (int i = t; i < H1 * 2 * KSEL; i += 128) {       // g_flat: feat[52 + h*20 + j]
        const int h = i / (2 * KSEL);
        const int j = i % (2 * KSEL);
        feat[2 * KSEL + H1 + i] = out_ws[((long)b * NTILE + ssel[j]) * H1 + h];
    }
    __syncthreads();
    if (t < H1) {                                        // avg over the 20 gathered rows
        float acc = 0.f;
        for (int j = 0; j < 2 * KSEL; ++j) acc += feat[2 * KSEL + H1 + t * 2 * KSEL + j];
        feat[2 * KSEL + t] = acc * (1.0f / (2 * KSEL));
    }
    __syncthreads();
    if (t < HF) {
        float acc = bc1[t];
        for (int j = 0; j < FEAT; ++j) acc += Wc1[t * FEAT + j] * feat[j];
        z1[t] = acc > 0.f ? acc : 0.f;
    }
    __syncthreads();
    if (t < HF) {
        float acc = bc2[t];
#pragma unroll
        for (int j = 0; j < HF; ++j) acc += Wc2[t * HF + j] * z1[j];
        z2[t] = acc > 0.f ? acc : 0.f;
    }
    __syncthreads();
    if (t == 0) {
        float acc = bc3[0];
#pragma unroll
        for (int j = 0; j < HF; ++j) acc += Wc3[j] * z2[j];
        out[b] = 1.0f / (1.0f + __expf(-acc));
    }
}

// ---------------------------------------------------------------------------
extern "C" void kernel_launch(void* const* d_in, const int* in_sizes, int n_in,
                              void* d_out, int out_size, void* d_ws, size_t ws_size,
                              hipStream_t stream) {
    const float* x   = (const float*)d_in[0];
    const float* W1  = (const float*)d_in[1];
    const float* b1  = (const float*)d_in[2];
    const float* W2  = (const float*)d_in[3];
    const float* b2  = (const float*)d_in[4];
    const float* W3  = (const float*)d_in[5];
    const float* b3  = (const float*)d_in[6];
    const float* Wsc = (const float*)d_in[7];
    const float* bsc = (const float*)d_in[8];
    const float* Wc1 = (const float*)d_in[9];
    const float* bc1 = (const float*)d_in[10];
    const float* Wc2 = (const float*)d_in[11];
    const float* bc2 = (const float*)d_in[12];
    const float* Wc3 = (const float*)d_in[13];
    const float* bc3 = (const float*)d_in[14];

    const long rows = (long)BATCH * NTILE;               // 160000
    float* out_ws    = (float*)d_ws;                     // [rows, 32]
    float* scores_ws = out_ws + rows * H1;               // [rows]
    int*   sel_ws    = (int*)(scores_ws + rows);         // [8, 20]

    conan_stage1<<<dim3(rows / ROWS_PER_BLOCK), dim3(256), 0, stream>>>(
        x, W1, b1, W2, b2, W3, b3, Wsc, bsc, out_ws, scores_ws);

    conan_select<<<dim3(BATCH), dim3(256), 0, stream>>>(scores_ws, sel_ws);

    conan_classify<<<dim3(BATCH), dim3(128), 0, stream>>>(
        out_ws, scores_ws, sel_ws, Wc1, bc1, Wc2, bc2, Wc3, bc3, (float*)d_out);
}
